// LSTM_32478542693143
// MI455X (gfx1250) — matrix operand order repaired and checked by reading, not compiled, not run on hardware
//
#include <hip/hip_runtime.h>

// ---------------------------------------------------------------------------
// Types
// ---------------------------------------------------------------------------
typedef _Float16 half8  __attribute__((ext_vector_type(8)));
typedef _Float16 v16h   __attribute__((ext_vector_type(16)));
typedef float    v8f    __attribute__((ext_vector_type(8)));

#define B_   16
#define T_   2048
#define K_   512      // D == H == 512
#define G4H  2048     // 4*H
#define NWG  16       // workgroups in the persistent recurrence kernel

// ---------------------------------------------------------------------------
// f32 -> f16 conversion
// ---------------------------------------------------------------------------
__global__ void cvt_f32_to_f16(const float* __restrict__ in,
                               _Float16* __restrict__ out, int n) {
  int i = blockIdx.x * blockDim.x + threadIdx.x;
  if (i < n) out[i] = (_Float16)in[i];
}

// ---------------------------------------------------------------------------
// Init scratch state (h ping-pong buffer + global barrier counter)
// ---------------------------------------------------------------------------
__global__ void init_state(_Float16* hbuf, unsigned* bar) {
  int i = threadIdx.x;
  for (int k = i; k < 2 * B_ * K_; k += blockDim.x) hbuf[k] = (_Float16)0.0f;
  if (i < 16) bar[i] = 0u;
}

// ---------------------------------------------------------------------------
// Phase A: WX = Xf16 @ Wf16^T + bias     (M x 2048, K=512)
// block = 256 threads (8 waves). WG tile: 128(M) x 64(N).
// wave w owns rows [m0+16w, +16), 4 N-tiles of 16.
// ---------------------------------------------------------------------------
__global__ void __launch_bounds__(256)
gemm_wx(const _Float16* __restrict__ A,   // [M][512]
        const _Float16* __restrict__ W,   // [2048][512]
        const float*    __restrict__ bias,// [2048]
        float* __restrict__ out,          // [M][2048]
        int M) {
  const int wave = threadIdx.x >> 5;
  const int lane = threadIdx.x & 31;
  const int m0   = blockIdx.x * 128 + wave * 16;
  const int n0   = blockIdx.y * 64;
  const int lm   = lane & 15;   // row within tile (A) / col within tile (B)
  const int g    = lane >> 4;   // lane group

  v8f acc0 = {}, acc1 = {}, acc2 = {}, acc3 = {};

  for (int kt = 0; kt < 16; ++kt) {
    // A fragment: row m0+lm, chunks at K = kt*32 + g*16 and +8
    const _Float16* ap = A + (size_t)(m0 + lm) * K_ + kt * 32 + g * 16;
    half8 a0 = *(const half8*)ap;
    half8 a1 = *(const half8*)(ap + 8);
    v16h  Af = __builtin_shufflevector(a0, a1, 0,1,2,3,4,5,6,7,8,9,10,11,12,13,14,15);

#pragma unroll
    for (int j = 0; j < 4; ++j) {
      const _Float16* bp = W + (size_t)(n0 + j * 16 + lm) * K_ + kt * 32 + g * 16;
      half8 b0 = *(const half8*)bp;
      half8 b1 = *(const half8*)(bp + 8);
      v16h  Bf = __builtin_shufflevector(b0, b1, 0,1,2,3,4,5,6,7,8,9,10,11,12,13,14,15);
      v8f& acc = (j == 0) ? acc0 : (j == 1) ? acc1 : (j == 2) ? acc2 : acc3;
      acc = __builtin_amdgcn_wmma_f32_16x16x32_f16(false, Af, false, Bf,
                                                   (short)0, acc, false, false);
    }
  }

#pragma unroll
  for (int j = 0; j < 4; ++j) {
    v8f acc = (j == 0) ? acc0 : (j == 1) ? acc1 : (j == 2) ? acc2 : acc3;
    int  n  = n0 + j * 16 + lm;
    float bs = bias[n];
#pragma unroll
    for (int r = 0; r < 8; ++r) {
      int m = m0 + g * 8 + r;
      out[(size_t)m * G4H + n] = acc[r] + bs;
    }
  }
}

// ---------------------------------------------------------------------------
// Device-scope grid barrier for NWG co-resident workgroups
// ---------------------------------------------------------------------------
__device__ __forceinline__ void grid_barrier(unsigned* bar) {
  __threadfence();
  __syncthreads();
  if (threadIdx.x == 0) {
    unsigned old    = __hip_atomic_fetch_add(bar, 1u, __ATOMIC_ACQ_REL,
                                             __HIP_MEMORY_SCOPE_AGENT);
    unsigned target = (old & ~(unsigned)(NWG - 1)) + NWG;
    while (__hip_atomic_load(bar, __ATOMIC_ACQUIRE,
                             __HIP_MEMORY_SCOPE_AGENT) < target) {
      __builtin_amdgcn_s_sleep(1);
    }
  }
  __syncthreads();
}

// ---------------------------------------------------------------------------
// Async copy global -> LDS (CDNA5 GLOBAL_LOAD_ASYNC_TO_LDS, ASYNCcnt-tracked).
// lds_off: byte offset within the workgroup's LDS allocation (dynamic LDS
// starts at 0 here: no static __shared__ in this kernel).
// ---------------------------------------------------------------------------
__device__ __forceinline__ void async_g2l_b128(unsigned lds_off,
                                               const void* gptr) {
  unsigned long long ga = (unsigned long long)(uintptr_t)gptr;
  asm volatile("global_load_async_to_lds_b128 %0, %1, off"
               :: "v"(lds_off), "v"(ga) : "memory");
}

__device__ __forceinline__ void wait_asynccnt0() {
#if __has_builtin(__builtin_amdgcn_s_wait_asynccnt)
  __builtin_amdgcn_s_wait_asynccnt(0);
#else
  asm volatile("s_wait_asynccnt 0x0" ::: "memory");
#endif
}

// ---------------------------------------------------------------------------
// Phase B: persistent recurrence.  grid = NWG blocks x 256 threads.
// WG `wg` owns h channels [wg*32, wg*32+32)  ->  gate cols {q*512 + wg*32 + c}.
// Wave w (0..7): gate chunk q = w>>1, col sub-block (w&1)*16.
// Dynamic LDS: Wu frags (128KB) | h stage (16KB) | gates (8KB) | c (2KB)
// ---------------------------------------------------------------------------
#define SM_U   0
#define SM_H   (128 * 1024)
#define SM_G   (144 * 1024)
#define SM_C   (152 * 1024)
#define SM_TOT (154 * 1024)

__global__ void __launch_bounds__(256)
lstm_recurrent(const float*    __restrict__ wx,    // [B][T][2048]
               const _Float16* __restrict__ U,     // [2048][512] f16
               const float*    __restrict__ ub,    // [2048]
               _Float16*       hbuf,               // [2][B][512] ping-pong
               _Float16*       hseq,               // [B][T][512] or null
               float*          outf32,             // [B][T][512] or null
               unsigned*       bar) {
  extern __shared__ char smem[];
  _Float16* sU = (_Float16*)(smem + SM_U);
  _Float16* sH = (_Float16*)(smem + SM_H);
  float*    sG = (float*)(smem + SM_G);
  float*    sC = (float*)(smem + SM_C);

  const int wg   = blockIdx.x;
  const int tid  = threadIdx.x;
  const int wave = tid >> 5;
  const int lane = tid & 31;
  const int lm   = lane & 15;
  const int g    = lane >> 4;
  const int gq   = wave >> 1;                         // 0..3 : i,f,g,o
  const int colbase = gq * 512 + wg * 32 + (wave & 1) * 16;
  const int gcol    = colbase + lm;                   // this lane's gate col
  const int gidx    = gq * 32 + (wave & 1) * 16 + lm; // col within local gates

  // --- pack this WG's Wu slice into LDS in B-fragment order -----------------
  for (int kt = 0; kt < 16; ++kt) {
    const _Float16* src = U + (size_t)gcol * K_ + kt * 32 + g * 16;
    _Float16* dst = sU + ((size_t)(wave * 16 + kt) * 32 + lane) * 16;
    *(half8*)dst       = *(const half8*)src;
    *(half8*)(dst + 8) = *(const half8*)(src + 8);
  }
  for (int i = tid; i < B_ * 32; i += 256) sC[i] = 0.0f;
  const float bs = ub[gcol];
  __syncthreads();

  for (int t = 0; t < T_; ++t) {
    // --- stage h(t-1) into LDS via async DMA (16 KB, cooperative) -----------
    const _Float16* hsrc = hbuf + (size_t)(t & 1) * B_ * K_;
#pragma unroll
    for (int i = tid * 8; i < B_ * K_; i += 256 * 8)
      async_g2l_b128((unsigned)(SM_H + i * 2), hsrc + i);
    if (t + 1 < T_)
      __builtin_prefetch(wx + ((size_t)(g * 8) * T_ + (t + 1)) * G4H + gcol, 0, 1);
    wait_asynccnt0();
    __syncthreads();

    // --- gates slice = h @ Wu^T  (16 x 16 tile per wave, K = 512) -----------
    v8f acc = {};
    for (int kt = 0; kt < 16; ++kt) {
      const _Float16* ap = sH + (size_t)lm * K_ + kt * 32 + g * 16;
      half8 a0 = *(const half8*)ap;
      half8 a1 = *(const half8*)(ap + 8);
      v16h  Af = __builtin_shufflevector(a0, a1, 0,1,2,3,4,5,6,7,8,9,10,11,12,13,14,15);
      const _Float16* bp = sU + ((size_t)(wave * 16 + kt) * 32 + lane) * 16;
      half8 b0 = *(const half8*)bp;
      half8 b1 = *(const half8*)(bp + 8);
      v16h  Bf = __builtin_shufflevector(b0, b1, 0,1,2,3,4,5,6,7,8,9,10,11,12,13,14,15);
      acc = __builtin_amdgcn_wmma_f32_16x16x32_f16(false, Af, false, Bf,
                                                   (short)0, acc, false, false);
    }

    // --- add wx + ub, exchange through LDS ----------------------------------
#pragma unroll
    for (int r = 0; r < 8; ++r) {
      int b = g * 8 + r;
      float v = acc[r] + wx[((size_t)b * T_ + t) * G4H + gcol] + bs;
      sG[b * 128 + gidx] = v;
    }
    __syncthreads();

    // --- elementwise LSTM cell update (512 elems / WG) ----------------------
    for (int e = tid; e < B_ * 32; e += 256) {
      int b  = e >> 5;
      int jl = e & 31;
      float iv = sG[b * 128 +       jl];
      float fv = sG[b * 128 +  32 + jl];
      float gv = sG[b * 128 +  64 + jl];
      float ov = sG[b * 128 +  96 + jl];
      iv = 1.0f / (1.0f + __expf(-iv));
      fv = 1.0f / (1.0f + __expf(-fv));
      gv = tanhf(gv);
      ov = 1.0f / (1.0f + __expf(-ov));
      float c = fv * sC[e] + iv * gv;
      sC[e] = c;
      float h = ov * fmaxf(c, 0.0f);          // nonlinearity = relu
      int j = wg * 32 + jl;
      hbuf[(size_t)((t + 1) & 1) * B_ * K_ + b * K_ + j] = (_Float16)h;
      if (hseq)   hseq[((size_t)b * T_ + t) * K_ + j] = (_Float16)h;
      if (outf32) outf32[((size_t)b * T_ + t) * K_ + j] = h;
    }

    grid_barrier(bar);   // release h(t) to all WGs
  }
}

// ---------------------------------------------------------------------------
// Host side
// ---------------------------------------------------------------------------
static inline size_t align_up(size_t x, size_t a) { return (x + a - 1) & ~(a - 1); }

extern "C" void kernel_launch(void* const* d_in, const int* in_sizes, int n_in,
                              void* d_out, int out_size, void* d_ws, size_t ws_size,
                              hipStream_t stream) {
  const float* x   = (const float*)d_in[0];
  const float* w0  = (const float*)d_in[1];
  const float* b0  = (const float*)d_in[2];
  const float* u0  = (const float*)d_in[3];
  const float* ub0 = (const float*)d_in[4];
  const float* w1  = (const float*)d_in[5];
  const float* b1  = (const float*)d_in[6];
  const float* u1  = (const float*)d_in[7];
  const float* ub1 = (const float*)d_in[8];
  float* out = (float*)d_out;

  const int M    = B_ * T_;          // 32768
  const int NX   = M * K_;           // x / h-seq elements
  const int NWt  = G4H * K_;         // one weight matrix

  // workspace carve-up
  char* p = (char*)d_ws;
  size_t off = 0;
  _Float16* xbf  = (_Float16*)(p + off); off = align_up(off + (size_t)NX  * 2, 256);
  _Float16* w0bf = (_Float16*)(p + off); off = align_up(off + (size_t)NWt * 2, 256);
  _Float16* u0bf = (_Float16*)(p + off); off = align_up(off + (size_t)NWt * 2, 256);
  _Float16* w1bf = (_Float16*)(p + off); off = align_up(off + (size_t)NWt * 2, 256);
  _Float16* u1bf = (_Float16*)(p + off); off = align_up(off + (size_t)NWt * 2, 256);
  float*    wx   = (float*)   (p + off); off = align_up(off + (size_t)M * G4H * 4, 256);
  _Float16* hbuf = (_Float16*)(p + off); off = align_up(off + (size_t)2 * B_ * K_ * 2, 256);
  unsigned* bar  = (unsigned*)(p + off); off = align_up(off + 64, 256);
  (void)ws_size; (void)n_in; (void)in_sizes; (void)out_size;

  // allow 154KB dynamic LDS for the persistent kernel
  hipFuncSetAttribute(reinterpret_cast<const void*>(lstm_recurrent),
                      hipFuncAttributeMaxDynamicSharedMemorySize, SM_TOT);

  // f32 -> f16 conversions
  cvt_f32_to_f16<<<(NX  + 255) / 256, 256, 0, stream>>>(x,  xbf,  NX);
  cvt_f32_to_f16<<<(NWt + 255) / 256, 256, 0, stream>>>(w0, w0bf, NWt);
  cvt_f32_to_f16<<<(NWt + 255) / 256, 256, 0, stream>>>(u0, u0bf, NWt);
  cvt_f32_to_f16<<<(NWt + 255) / 256, 256, 0, stream>>>(w1, w1bf, NWt);
  cvt_f32_to_f16<<<(NWt + 255) / 256, 256, 0, stream>>>(u1, u1bf, NWt);

  dim3 ggrid(M / 128, G4H / 64);

  // ---- layer 0 ----
  gemm_wx<<<ggrid, 256, 0, stream>>>(xbf, w0bf, b0, wx, M);
  init_state<<<1, 256, 0, stream>>>(hbuf, bar);
  lstm_recurrent<<<NWG, 256, SM_TOT, stream>>>(wx, u0bf, ub0, hbuf,
                                               /*hseq=*/xbf, /*outf32=*/nullptr, bar);

  // ---- layer 1 (input = layer-0 h sequence, written into xbf) ----
  gemm_wx<<<ggrid, 256, 0, stream>>>(xbf, w1bf, b1, wx, M);
  init_state<<<1, 256, 0, stream>>>(hbuf, bar);
  lstm_recurrent<<<NWG, 256, SM_TOT, stream>>>(wx, u1bf, ub1, hbuf,
                                               /*hseq=*/nullptr, /*outf32=*/out, bar);
}
